// GCNIIdense_model_28140625724052
// MI455X (gfx1250) — matrix operand order repaired and checked by reading, not compiled, not run on hardware
//
#include <hip/hip_runtime.h>
#include <cstddef>
#include <cstdint>

// ---------------------------------------------------------------------------
// Model constants (match reference)
// ---------------------------------------------------------------------------
#define N_NODES 32768
#define HID     512
#define OUTD    128
#define ALPHA_C 0.2f

typedef __bf16 bf16;
typedef bf16  v8bf  __attribute__((ext_vector_type(8)));
typedef bf16  v16bf __attribute__((ext_vector_type(16)));
typedef float v8f   __attribute__((ext_vector_type(8)));

// float -> bf16, round-to-nearest-even (matches HW RNE used by WMMA)
__device__ __forceinline__ unsigned short f2bf(float f) {
    unsigned int u = __float_as_uint(f);
    unsigned int r = u + 0x7FFFu + ((u >> 16) & 1u);
    return (unsigned short)(r >> 16);
}
// pack two floats into packed bf16x2 dword (lo in [15:0], hi in [31:16])
__device__ __forceinline__ unsigned int f2bf2(float lo, float hi) {
    return (unsigned int)f2bf(lo) | ((unsigned int)f2bf(hi) << 16);
}

// ---------------------------------------------------------------------------
// Staging helpers (128x128 tile, K-step 64). LDS:
//   A: 128 rows x 64 bf16, row-major        (Asu = dword view)
//   B: 128 cols x 64 bf16, K-contig per col (Bsu = dword view)
// ---------------------------------------------------------------------------
__device__ __forceinline__ void stage_a_fast(const float* __restrict__ A, int lda,
                                             int mBlock, int k0, int tid,
                                             unsigned int* Asu) {
    const int arow = tid >> 1;        // 0..127
    const int acol = (tid & 1) * 32;  // 0 or 32
    const float4* ap = (const float4*)(A + (size_t)(mBlock + arow) * lda + k0 + acol);
    unsigned int* dst = &Asu[arow * 32 + (acol >> 1)];
#pragma unroll
    for (int j = 0; j < 4; ++j) {
        const float4 x0 = ap[2 * j];
        const float4 x1 = ap[2 * j + 1];
        uint4 p;
        p.x = f2bf2(x0.x, x0.y);
        p.y = f2bf2(x0.z, x0.w);
        p.z = f2bf2(x1.x, x1.y);
        p.w = f2bf2(x1.z, x1.w);
        ((uint4*)dst)[j] = p;
    }
}

// clamped scalar loads + select-to-zero (no exec-masked branches); handles K tail
__device__ __forceinline__ void stage_a_slow(const float* __restrict__ A, int lda,
                                             int mBlock, int k0, int K, int tid,
                                             unsigned short* Ass) {
    const int arow = tid >> 1;
    const int acol = (tid & 1) * 32;
    const size_t rbase = (size_t)(mBlock + arow) * lda;
#pragma unroll
    for (int i = 0; i < 32; ++i) {
        const int kg = k0 + acol + i;
        const int kc = kg < K ? kg : (K - 1);
        const float v = A[rbase + kc];
        Ass[arow * 64 + acol + i] = f2bf(kg < K ? v : 0.0f);
    }
}

__device__ __forceinline__ void stage_b_fast(const float* __restrict__ W, int ldb,
                                             int nBlock, int k0, int tid,
                                             unsigned int* Bsu) {
    const int fkk = (tid & 31) * 2;   // 0..62  (K pair)
    const int fnn = (tid >> 5) * 16;  // 0..112 (16 columns)
    const float* b0 = W + (size_t)(k0 + fkk) * ldb + nBlock + fnn;
    const float* b1 = b0 + ldb;
#pragma unroll
    for (int j = 0; j < 4; ++j) {
        const float4 q0 = ((const float4*)b0)[j];
        const float4 q1 = ((const float4*)b1)[j];
        const int nb = fnn + j * 4;
        Bsu[(nb + 0) * 32 + (fkk >> 1)] = f2bf2(q0.x, q1.x);
        Bsu[(nb + 1) * 32 + (fkk >> 1)] = f2bf2(q0.y, q1.y);
        Bsu[(nb + 2) * 32 + (fkk >> 1)] = f2bf2(q0.z, q1.z);
        Bsu[(nb + 3) * 32 + (fkk >> 1)] = f2bf2(q0.w, q1.w);
    }
}

__device__ __forceinline__ void stage_b_slow(const float* __restrict__ W, int ldb,
                                             int nBlock, int k0, int K, int tid,
                                             unsigned short* Bss) {
    const int brow = tid >> 2;        // 0..63 (K within tile)
    const int bcol = (tid & 3) * 32;  // 0..96 (N within tile)
    const int kg = k0 + brow;
    const int kc = kg < K ? kg : (K - 1);
    const float* bp = W + (size_t)kc * ldb + nBlock + bcol;
#pragma unroll
    for (int i = 0; i < 32; ++i) {
        const float v = bp[i];
        Bss[(bcol + i) * 64 + brow] = f2bf(kg < K ? v : 0.0f);
    }
}

// 2 k-slices x (2 M x 4 N) = 16 WMMA tiles; A/B fragments use contiguous k = 16*(lane/16) + s
__device__ __forceinline__ void compute64(const unsigned short* Ass,
                                          const unsigned short* Bss,
                                          int wm, int wn, int lhalf, int lmod,
                                          v8f (&acc)[2][4]) {
#pragma unroll
    for (int kk0 = 0; kk0 < 64; kk0 += 32) {
        // A: lane L (M=L%16): elems 0..15 = K kb+0..15,
        // kb = (L>=16)?16:0
        v16bf afrag[2];
#pragma unroll
        for (int mt = 0; mt < 2; ++mt) {
            const int row = wm * 32 + mt * 16 + lmod;
            const v8bf lo = *(const v8bf*)&Ass[row * 64 + kk0 + lhalf * 16];
            const v8bf hi = *(const v8bf*)&Ass[row * 64 + kk0 + lhalf * 16 + 8];
#pragma unroll
            for (int i = 0; i < 8; ++i) { afrag[mt][i] = lo[i]; afrag[mt][8 + i] = hi[i]; }
        }
        // B: lane L (N=L%16) holds K kk0+0..15 (lanes 0-15) / kk0+16..31 (lanes 16-31)
#pragma unroll
        for (int nt = 0; nt < 4; ++nt) {
            const int ncol = wn * 64 + nt * 16 + lmod;
            const v16bf bfrag = *(const v16bf*)&Bss[ncol * 64 + kk0 + lhalf * 16];
#pragma unroll
            for (int mt = 0; mt < 2; ++mt) {
                acc[mt][nt] = __builtin_amdgcn_wmma_f32_16x16x32_bf16(
                    false, afrag[mt], false, bfrag, (short)0, acc[mt][nt], false, false);
            }
        }
    }
}

// ---------------------------------------------------------------------------
// WMMA GEMM:  C = epilogue( A[M,K] @ W[K,N] )  with bf16 inputs, f32 accum.
//   epilogue: v = acc (+bias[n]) (+add_pre[m,n]); if relu v=max(v,0); (+add_post[m,n])
// Block = 256 threads (8 wave32), tile 128x128, K-step 64.
// Waves 4(M) x 2(N); each wave owns 2x4 16x16 accumulator tiles.
// Main loop = full K chunks only (branch-free fast staging);
// partial K tail (K=78/54 input GEMMs only) peeled after the loop.
// LDA_VEC template removes the unaligned-A path from the main loop entirely.
// ---------------------------------------------------------------------------
template <bool LDA_VEC>
__global__ __launch_bounds__(256) void k_gemm_bf16(
    const float* __restrict__ A, int lda,
    const float* __restrict__ W, int ldb,
    float* __restrict__ C, int ldc,
    const float* __restrict__ bias,
    const float* __restrict__ add_pre, int ldpre,
    const float* __restrict__ add_post, int ldpost,
    int K, int relu)
{
    __shared__ __align__(32) unsigned int Asu[128 * 32];   // 128 x 64 bf16
    __shared__ __align__(32) unsigned int Bsu[128 * 32];   // 128 x 64 bf16 (transposed)
    unsigned short* Ass = (unsigned short*)Asu;
    unsigned short* Bss = (unsigned short*)Bsu;

    const int tid   = threadIdx.x;
    const int lane  = tid & 31;
    const int wave  = tid >> 5;
    const int wm    = wave & 3;    // wave M tile: rows [wm*32, wm*32+32)
    const int wn    = wave >> 2;   // wave N tile: cols [wn*64, wn*64+64)
    const int lhalf = lane >> 4;
    const int lmod  = lane & 15;

    const int nBlock = blockIdx.x * 128;
    const int mBlock = blockIdx.y * 128;

    v8f acc[2][4];
#pragma unroll
    for (int mt = 0; mt < 2; ++mt)
#pragma unroll
        for (int nt = 0; nt < 4; ++nt)
#pragma unroll
            for (int r = 0; r < 8; ++r) acc[mt][nt][r] = 0.0f;

    const int KtFull = K >> 6;

    for (int kt = 0; kt < KtFull; ++kt) {
        const int k0 = kt << 6;
        if (LDA_VEC) stage_a_fast(A, lda, mBlock, k0, tid, Asu);
        else         stage_a_slow(A, lda, mBlock, k0, K, tid, Ass);
        stage_b_fast(W, ldb, nBlock, k0, tid, Bsu);
        // speculative prefetch of next slice (dropped harmlessly past end)
        __builtin_prefetch(A + (size_t)(mBlock + (tid >> 1)) * lda + (k0 + 64), 0, 3);
        __builtin_prefetch(W + (size_t)(k0 + 64 + (tid & 31) * 2) * ldb + nBlock, 0, 3);
        __syncthreads();
        compute64(Ass, Bss, wm, wn, lhalf, lmod, acc);
        __syncthreads();
    }
    if (K & 63) {                      // peeled partial tail (input GEMMs only)
        const int k0 = KtFull << 6;
        stage_a_slow(A, lda, mBlock, k0, K, tid, Ass);
        stage_b_slow(W, ldb, nBlock, k0, K, tid, Bss);
        __syncthreads();
        compute64(Ass, Bss, wm, wn, lhalf, lmod, acc);
    }

    // ---- epilogue (C/D layout: VGPR r -> M = lhalf*8 + r, N = lane%16)
#pragma unroll
    for (int mt = 0; mt < 2; ++mt) {
#pragma unroll
        for (int nt = 0; nt < 4; ++nt) {
            const int n = nBlock + wn * 64 + nt * 16 + lmod;
#pragma unroll
            for (int r = 0; r < 8; ++r) {
                const int m = mBlock + wm * 32 + mt * 16 + lhalf * 8 + r;
                float v = acc[mt][nt][r];
                if (bias)     v += bias[n];
                if (add_pre)  v += add_pre[(size_t)m * ldpre + n];
                if (relu)     v = fmaxf(v, 0.0f);
                if (add_post) v += add_post[(size_t)m * ldpost + n];
                C[(size_t)m * ldc + n] = v;
            }
        }
    }
}

// ---------------------------------------------------------------------------
// Graph-norm + SpMM helper kernels (bandwidth bound; VALU/atomic path)
// ---------------------------------------------------------------------------
__global__ void k_fill(float* __restrict__ p, float v, int n) {
    int i = blockIdx.x * blockDim.x + threadIdx.x;
    if (i < n) p[i] = v;
}

__global__ void k_deg_edges(const int* __restrict__ col, float* __restrict__ deg, int E) {
    int e = blockIdx.x * blockDim.x + threadIdx.x;
    if (e < E) atomicAdd(&deg[col[e]], 1.0f);
}

__global__ void k_dis(float* __restrict__ deg, int n) {
    int i = blockIdx.x * blockDim.x + threadIdx.x;
    if (i < n) { float d = deg[i]; deg[i] = d > 0.0f ? rsqrtf(d) : 0.0f; }
}

// agg[i,f] = dis[i]^2 * prev[i,f]   (self-loop term; also initializes agg)
__global__ void k_spmm_init(const float* __restrict__ prev, const float* __restrict__ dis,
                            float* __restrict__ agg, int n) {
    int i = blockIdx.x * blockDim.x + threadIdx.x;
    if (i < n) { float s = dis[i >> 9]; agg[i] = s * s * prev[i]; }  // HID=512
}

// edge scatter: agg[col,:] += dis[row]*dis[col] * prev[row,:]
__global__ void k_spmm_edges(const int* __restrict__ row, const int* __restrict__ col,
                             const float* __restrict__ dis, const float* __restrict__ prev,
                             float* __restrict__ agg, int E) {
    const int e = blockIdx.x;
    const int f = (blockIdx.y << 8) + threadIdx.x;
    const int r = row[e], c = col[e];              // uniform -> scalar loads
    const float nrm = dis[r] * dis[c];
    atomicAdd(&agg[(size_t)c * HID + f], nrm * prev[(size_t)r * HID + f]);
}

// support = (1-alpha)*agg + alpha*h0
__global__ void k_support(const float* __restrict__ agg, const float* __restrict__ h0,
                          float* __restrict__ supp, int n) {
    int i = blockIdx.x * blockDim.x + threadIdx.x;
    if (i < n) supp[i] = (1.0f - ALPHA_C) * agg[i] + ALPHA_C * h0[i];
}

// out[i] = X[i,:512] . w + b   (512->1 head; trivially bandwidth bound)
__global__ void k_head(const float* __restrict__ X, const float* __restrict__ w,
                       const float* __restrict__ b, float* __restrict__ out, int n) {
    int i = blockIdx.x * blockDim.x + threadIdx.x;
    if (i >= n) return;
    const float4* xp = (const float4*)(X + (size_t)i * 512);
    const float4* wp = (const float4*)w;
    float s = 0.0f;
#pragma unroll 4
    for (int k = 0; k < 128; ++k) {
        float4 a = xp[k], ww = wp[k];
        s = fmaf(a.x, ww.x, fmaf(a.y, ww.y, fmaf(a.z, ww.z, fmaf(a.w, ww.w, s))));
    }
    out[i] = s + b[0];
}

// ---------------------------------------------------------------------------
// Host-side sequencing
// ---------------------------------------------------------------------------
static inline void launch_gemm(const float* A, int lda, const float* W, int ldb,
                               float* C, int ldc, const float* bias,
                               const float* pre, int ldpre, const float* post, int ldpost,
                               int M, int Nc, int K, int relu, hipStream_t s) {
    dim3 grid(Nc / 128, M / 128);
    if ((lda & 3) == 0)
        k_gemm_bf16<true><<<grid, 256, 0, s>>>(A, lda, W, ldb, C, ldc, bias,
                                               pre, ldpre, post, ldpost, K, relu);
    else
        k_gemm_bf16<false><<<grid, 256, 0, s>>>(A, lda, W, ldb, C, ldc, bias,
                                                pre, ldpre, post, ldpost, K, relu);
}

static void run_branch(const float* x, int F, const int* ei, int E,
                       const float* w_in, const float* b_in, const float* ws3,
                       const float* w_out, const float* b_out,
                       float* dis, float* h0, float* prev, float* agg, float* supp,
                       float* xc, int xcOff, hipStream_t s) {
    const int n  = N_NODES;
    const int nf = n * HID;
    const int T  = 256;

    // deg (with self loops) -> dis = rsqrt(deg)
    k_fill<<<(n + T - 1) / T, T, 0, s>>>(dis, 1.0f, n);
    k_deg_edges<<<(E + T - 1) / T, T, 0, s>>>(ei + E, dis, E);   // col = second row
    k_dis<<<(n + T - 1) / T, T, 0, s>>>(dis, n);

    // h0 = relu(x @ w_in + b_in)
    launch_gemm(x, F, w_in, HID, h0, HID, b_in, nullptr, 0, nullptr, 0, n, HID, F, 1, s);
    hipMemcpyAsync(prev, h0, (size_t)nf * sizeof(float), hipMemcpyDeviceToDevice, s);

    float* pv = prev;
    float* ag = agg;
    for (int l = 0; l < 3; ++l) {
        k_spmm_init<<<(nf + T - 1) / T, T, 0, s>>>(pv, dis, ag, nf);
        dim3 ge(E, HID / T);
        k_spmm_edges<<<ge, T, 0, s>>>(ei, ei + E, dis, pv, ag, E);
        k_support<<<(nf + T - 1) / T, T, 0, s>>>(ag, h0, supp, nf);
        // new_prev = relu(supp + supp@W_l) + pv  -> written into ag, then swap
        launch_gemm(supp, HID, ws3 + (size_t)l * HID * HID, HID, ag, HID,
                    nullptr, supp, HID, pv, HID, n, HID, HID, 1, s);
        float* t = pv; pv = ag; ag = t;
    }
    // branch output straight into concat buffer (ldc = 256, column offset)
    launch_gemm(pv, HID, w_out, OUTD, xc + xcOff, 2 * OUTD, b_out,
                nullptr, 0, nullptr, 0, n, OUTD, HID, 0, s);
}

extern "C" void kernel_launch(void* const* d_in, const int* in_sizes, int n_in,
                              void* d_out, int out_size, void* d_ws, size_t ws_size,
                              hipStream_t stream) {
    (void)n_in; (void)out_size; (void)ws_size;
    const float* mol_x    = (const float*)d_in[0];
    const int*   mol_ei   = (const int*)  d_in[1];
    const float* pro_x    = (const float*)d_in[2];
    const int*   pro_ei   = (const int*)  d_in[3];
    const float* mol_w_in = (const float*)d_in[4];
    const float* mol_b_in = (const float*)d_in[5];
    const float* mol_ws   = (const float*)d_in[6];
    const float* mol_w_out= (const float*)d_in[7];
    const float* mol_b_out= (const float*)d_in[8];
    const float* pro_w_in = (const float*)d_in[9];
    const float* pro_b_in = (const float*)d_in[10];
    const float* pro_ws   = (const float*)d_in[11];
    const float* pro_w_out= (const float*)d_in[12];
    const float* pro_b_out= (const float*)d_in[13];
    const float* w_fc1    = (const float*)d_in[14];
    const float* b_fc1    = (const float*)d_in[15];
    const float* w_fc2    = (const float*)d_in[16];
    const float* b_fc2    = (const float*)d_in[17];
    const float* w_o      = (const float*)d_in[18];
    const float* b_o      = (const float*)d_in[19];

    const int F_MOL = in_sizes[0] / N_NODES;   // 78
    const int F_PRO = in_sizes[2] / N_NODES;   // 54
    const int E_MOL = in_sizes[1] / 2;
    const int E_PRO = in_sizes[3] / 2;

    // workspace layout (floats): dis | h0 | prev | agg | supp | xc
    float* ws   = (float*)d_ws;
    const size_t n = N_NODES;
    float* dis  = ws;                          // N
    float* h0   = dis  + n;                    // N*512
    float* prev = h0   + n * HID;              // N*512
    float* agg  = prev + n * HID;              // N*512
    float* supp = agg  + n * HID;              // N*512
    float* xc   = supp + n * HID;              // N*256
    float* fc1o = h0;                          // reuse h0+prev region (N*1024)
    float* fc2o = agg;                         // reuse (N*512)

    run_branch(mol_x, F_MOL, mol_ei, E_MOL, mol_w_in, mol_b_in, mol_ws,
               mol_w_out, mol_b_out, dis, h0, prev, agg, supp, xc, 0, stream);
    run_branch(pro_x, F_PRO, pro_ei, E_PRO, pro_w_in, pro_b_in, pro_ws,
               pro_w_out, pro_b_out, dis, h0, prev, agg, supp, xc, OUTD, stream);

    // MLP head
    launch_gemm(xc,   2 * OUTD, w_fc1, 1024, fc1o, 1024, b_fc1,
                nullptr, 0, nullptr, 0, (int)n, 1024, 2 * OUTD, 1, stream);
    launch_gemm(fc1o, 1024,     w_fc2, 512,  fc2o, 512,  b_fc2,
                nullptr, 0, nullptr, 0, (int)n, 512, 1024, 1, stream);
    k_head<<<((int)n + 255) / 256, 256, 0, stream>>>(fc2o, w_o, b_o, (float*)d_out, (int)n);
}
